// MixtureOfHeadsAttention_55284819034462
// MI455X (gfx1250) — matrix operand order repaired and checked by reading, not compiled, not run on hardware
//
#include <hip/hip_runtime.h>
#include <hip/hip_bf16.h>

// ---------------- problem constants ----------------
#define BB 8
#define TT 1024
#define DD 1024
#define HH 16
#define DH 64
#define KK 4

// ---------------- types ----------------
typedef __attribute__((ext_vector_type(16))) __bf16         v16bf;
typedef __attribute__((ext_vector_type(8)))  float          v8f;
typedef __attribute__((ext_vector_type(8)))  unsigned short u16x8;
typedef __attribute__((ext_vector_type(16))) unsigned short u16x16;

union BF16Frag {
    u16x16 u;
    v16bf  bf;
    u16x8  h[2];
};

__device__ __forceinline__ unsigned short f2bf(float f) {
    unsigned int u = __float_as_uint(f);
    unsigned int r = (u + 0x7FFFu + ((u >> 16) & 1u)) >> 16;   // RNE
    return (unsigned short)r;
}

// A fragment (16x32 bf16, MxK): prow points at row (lane&15), element k0.
// a[0..15] = K = half*16 + 0..15    (32 contiguous bytes)
__device__ __forceinline__ v16bf load_a_frag(const unsigned short* prow, int half) {
    BF16Frag f;
    f.u = *(const u16x16*)(prow + half * 16);
    return f.bf;
}

// B fragment (32x16 bf16, KxN): pcol points at column (lane&15)'s K-contiguous
// storage, element k0 + half*16.  b[e] = B[half*16+e][col]  (32 contiguous bytes)
__device__ __forceinline__ v16bf load_b_frag(const unsigned short* pcol) {
    BF16Frag f;
    f.u = *(const u16x16*)pcol;
    return f.bf;
}

__device__ __forceinline__ v8f wmma_bf16(v16bf a, v16bf b, v8f c) {
    return __builtin_amdgcn_wmma_f32_16x16x32_bf16(false, a, false, b,
                                                   (short)0, c, false, false);
}

#define NEG_INF (-__builtin_inff())
// ln(10000)/32
#define ROPE_LC 0.287823136624255720f

// ---------------- conversion kernels ----------------
__global__ void conv_bf16_kernel(unsigned short* __restrict__ dst,
                                 const float* __restrict__ src, int n) {
    for (int i = blockIdx.x * blockDim.x + threadIdx.x; i < n;
         i += gridDim.x * blockDim.x)
        dst[i] = f2bf(src[i]);
}

// src (Hn, R, C) f32 -> dst (Hn, C, R) bf16
__global__ void transpose_conv_kernel(unsigned short* __restrict__ dst,
                                      const float* __restrict__ src,
                                      int Hn, int R, int C) {
    int tot = Hn * R * C;
    for (int i = blockIdx.x * blockDim.x + threadIdx.x; i < tot;
         i += gridDim.x * blockDim.x) {
        int h = i / (R * C);
        int rem = i - h * R * C;
        int c = rem / R;
        int r = rem - c * R;
        dst[i] = f2bf(src[(size_t)h * R * C + (size_t)r * C + c]);
    }
}

__global__ void zero_kernel(float* p, int n) {
    int i = blockIdx.x * blockDim.x + threadIdx.x;
    if (i < n) p[i] = 0.0f;
}

// ---------------- router: x @ rp^T, rope, mean-pool ----------------
// grid (8 tiles, B batches), 256 threads (8 waves); tile = 128 rows x 64 cols
__global__ void router_gemm_rope_pool(const unsigned short* __restrict__ xb,
                                      const unsigned short* __restrict__ rpb,
                                      const int* __restrict__ positions,
                                      float* __restrict__ pooled) {
    const int b    = blockIdx.y;
    const int tile = blockIdx.x;
    const int w    = threadIdx.x >> 5;
    const int lane = threadIdx.x & 31;
    const int col  = lane & 15;
    const int half = lane >> 4;
    const int row0 = tile * 128 + w * 16;

    const unsigned short* xrow = xb + ((size_t)b * TT + row0 + col) * DD;

    v8f acc[4] = {};
    for (int k0 = 0; k0 < DD; k0 += 32) {
        __builtin_prefetch(xrow + k0 + 256, 0, 3);   // near-cache (WGP) prefetch
        v16bf a = load_a_frag(xrow + k0, half);
#pragma unroll
        for (int nt = 0; nt < 4; ++nt) {
            v16bf bb = load_b_frag(rpb + (size_t)(nt * 16 + col) * DD + k0 + half * 16);
            acc[nt] = wmma_bf16(a, bb, acc[nt]);
        }
    }

    float posr[8];
#pragma unroll
    for (int r = 0; r < 8; ++r) posr[r] = (float)positions[row0 + r + 8 * half];

#pragma unroll
    for (int p = 0; p < 2; ++p) {   // pairs (0,2) and (1,3): h and h+32
        int   h    = p * 16 + col;  // 0..31
        float invf = __expf(-(float)h * ROPE_LC);
        float s1 = 0.0f, s2 = 0.0f;
#pragma unroll
        for (int r = 0; r < 8; ++r) {
            float ang = posr[r] * invf;
            float cs = __cosf(ang), sn = __sinf(ang);
            float x1 = acc[p][r], x2 = acc[p + 2][r];
            s1 += x1 * cs - x2 * sn;
            s2 += x2 * cs + x1 * sn;
        }
        atomicAdd(&pooled[b * DH + h], s1);
        atomicAdd(&pooled[b * DH + h + 32], s2);
    }
}

// ---------------- router: logits, top-4, softmax ----------------
__global__ void router_topk(const float* __restrict__ pooled,
                            const float* __restrict__ router_w,
                            int* __restrict__ idx, float* __restrict__ rw) {
    int b = threadIdx.x;
    if (b >= BB) return;
    float pb[DH];
    for (int i = 0; i < DH; ++i) pb[i] = pooled[b * DH + i] * (1.0f / (float)TT);
    float logits[HH];
    for (int n = 0; n < HH; ++n) {
        float s = 0.0f;
        for (int i = 0; i < DH; ++i) s += pb[i] * router_w[n * DH + i];
        logits[n] = s;
    }
    bool used[HH];
    for (int n = 0; n < HH; ++n) used[n] = false;
    float vals[KK]; int ids[KK];
    for (int j = 0; j < KK; ++j) {
        float best = NEG_INF; int bi = 0;
        for (int n = 0; n < HH; ++n)
            if (!used[n] && logits[n] > best) { best = logits[n]; bi = n; }
        used[bi] = true; vals[j] = best; ids[j] = bi;
    }
    float mx = vals[0], den = 0.0f, e[KK];
    for (int j = 0; j < KK; ++j) { e[j] = __expf(vals[j] - mx); den += e[j]; }
    for (int j = 0; j < KK; ++j) {
        idx[b * KK + j] = ids[j];
        rw[b * KK + j]  = e[j] / den;
    }
}

// ---------------- QKV projection + rope ----------------
// grid (8 tiles, B*K), 256 threads; per block: 128 rows x 64 cols of q,k,v
__global__ void qkv_proj_rope(const unsigned short* __restrict__ xb,
                              const unsigned short* __restrict__ wqt,
                              const unsigned short* __restrict__ wkt,
                              const unsigned short* __restrict__ wvt,
                              const int* __restrict__ idx,
                              const int* __restrict__ positions,
                              unsigned short* __restrict__ qb,
                              unsigned short* __restrict__ kb,
                              unsigned short* __restrict__ vtb) {
    const int bk   = blockIdx.y;
    const int b    = bk >> 2;
    const int tile = blockIdx.x;
    const int hsel = idx[bk];
    const int w    = threadIdx.x >> 5;
    const int lane = threadIdx.x & 31;
    const int col  = lane & 15;
    const int half = lane >> 4;
    const int row0 = tile * 128 + w * 16;

    const unsigned short* xrow = xb + ((size_t)b * TT + row0 + col) * DD;
    const unsigned short* wq = wqt + (size_t)hsel * DH * DD;
    const unsigned short* wk = wkt + (size_t)hsel * DH * DD;
    const unsigned short* wv = wvt + (size_t)hsel * DH * DD;

    v8f aq[4] = {}, ak[4] = {}, av[4] = {};
    for (int k0 = 0; k0 < DD; k0 += 32) {
        __builtin_prefetch(xrow + k0 + 256, 0, 3);   // near-cache (WGP) prefetch
        v16bf a = load_a_frag(xrow + k0, half);
#pragma unroll
        for (int nt = 0; nt < 4; ++nt) {
            size_t cb = (size_t)(nt * 16 + col) * DD + k0 + half * 16;
            aq[nt] = wmma_bf16(a, load_b_frag(wq + cb), aq[nt]);
            ak[nt] = wmma_bf16(a, load_b_frag(wk + cb), ak[nt]);
            av[nt] = wmma_bf16(a, load_b_frag(wv + cb), av[nt]);
        }
    }

    float posr[8];
#pragma unroll
    for (int r = 0; r < 8; ++r) posr[r] = (float)positions[row0 + r + 8 * half];

#pragma unroll
    for (int p = 0; p < 2; ++p) {
        int   h    = p * 16 + col;
        float invf = __expf(-(float)h * ROPE_LC);
#pragma unroll
        for (int r = 0; r < 8; ++r) {
            float ang = posr[r] * invf;
            float cs = __cosf(ang), sn = __sinf(ang);
            float q1 = aq[p][r], q2 = aq[p + 2][r];
            aq[p][r]     = q1 * cs - q2 * sn;
            aq[p + 2][r] = q2 * cs + q1 * sn;
            float k1 = ak[p][r], k2 = ak[p + 2][r];
            ak[p][r]     = k1 * cs - k2 * sn;
            ak[p + 2][r] = k2 * cs + k1 * sn;
        }
    }

    unsigned short* qout = qb  + (size_t)bk * TT * DH;
    unsigned short* kout = kb  + (size_t)bk * TT * DH;
    unsigned short* vout = vtb + (size_t)bk * DH * TT;
#pragma unroll
    for (int nt = 0; nt < 4; ++nt) {
#pragma unroll
        for (int r = 0; r < 8; ++r) {
            int trow = row0 + r + 8 * half;
            int dcol = nt * 16 + col;
            qout[(size_t)trow * DH + dcol] = f2bf(aq[nt][r]);
            kout[(size_t)trow * DH + dcol] = f2bf(ak[nt][r]);
            vout[(size_t)dcol * TT + trow] = f2bf(av[nt][r]);   // V transposed
        }
    }
}

// ---------------- flash attention ----------------
// grid (8 query tiles, B*K), 256 threads; wave handles 16 query rows
__global__ void attn_kernel(const unsigned short* __restrict__ qb,
                            const unsigned short* __restrict__ kb,
                            const unsigned short* __restrict__ vtb,
                            unsigned short* __restrict__ ctxb) {
    __shared__ __align__(64) unsigned short lds_p[8][16 * 32];

    const int bk   = blockIdx.y;
    const int tile = blockIdx.x;
    const int w    = threadIdx.x >> 5;
    const int lane = threadIdx.x & 31;
    const int col  = lane & 15;
    const int half = lane >> 4;
    const int q0   = tile * 128 + w * 16;

    const unsigned short* qrow  = qb + (size_t)bk * TT * DH + (size_t)(q0 + col) * DH;
    const unsigned short* kbase = kb + (size_t)bk * TT * DH;
    const unsigned short* vbase = vtb + (size_t)bk * DH * TT;
    unsigned short* pbuf = lds_p[w];

    v16bf a0 = load_a_frag(qrow, half);        // q dims 0..31
    v16bf a1 = load_a_frag(qrow + 32, half);   // q dims 32..63

    v8f acc[4] = {};
    float m[8], l[8];
#pragma unroll
    for (int r = 0; r < 8; ++r) { m[r] = NEG_INF; l[r] = 0.0f; }

    const float scale = 0.125f;   // DH^-0.5

    for (int kt = 0; kt < q0 + 16; kt += 32) {   // wave-uniform causal bound
        v8f sc[2] = {};
#pragma unroll
        for (int j = 0; j < 2; ++j) {
            const unsigned short* kc =
                kbase + (size_t)(kt + j * 16 + col) * DH + half * 16;
            sc[j] = wmma_bf16(a0, load_b_frag(kc), sc[j]);
            sc[j] = wmma_bf16(a1, load_b_frag(kc + 32), sc[j]);
        }
        // scale + causal mask
#pragma unroll
        for (int j = 0; j < 2; ++j)
#pragma unroll
            for (int r = 0; r < 8; ++r) {
                int key = kt + j * 16 + col;
                int row = q0 + r + 8 * half;
                float v = sc[j][r] * scale;
                sc[j][r] = (key > row) ? NEG_INF : v;
            }
        // online softmax per row
#pragma unroll
        for (int r = 0; r < 8; ++r) {
            float tm = fmaxf(sc[0][r], sc[1][r]);
#pragma unroll
            for (int s = 8; s >= 1; s >>= 1) tm = fmaxf(tm, __shfl_xor(tm, s, 16));
            float mn   = fmaxf(m[r], tm);
            float corr = (mn == NEG_INF) ? 1.0f : __expf(m[r] - mn);
            float p0   = (mn == NEG_INF) ? 0.0f : __expf(sc[0][r] - mn);
            float p1   = (mn == NEG_INF) ? 0.0f : __expf(sc[1][r] - mn);
            pbuf[(r + 8 * half) * 32 + col]      = f2bf(p0);
            pbuf[(r + 8 * half) * 32 + 16 + col] = f2bf(p1);
            float ts = p0 + p1;
#pragma unroll
            for (int s = 8; s >= 1; s >>= 1) ts += __shfl_xor(ts, s, 16);
            l[r] = l[r] * corr + ts;
            m[r] = mn;
#pragma unroll
            for (int nt = 0; nt < 4; ++nt) acc[nt][r] *= corr;
        }
        // same-wave LDS RAW: DS ops are in-order; wait + compiler barrier
        asm volatile("s_wait_dscnt 0" ::: "memory");
        // P @ V  (A = P 16x32 from LDS, B = V^T tiles from global)
        v16bf ap = load_a_frag(pbuf + col * 32, half);
#pragma unroll
        for (int nt = 0; nt < 4; ++nt) {
            const unsigned short* vc =
                vbase + (size_t)(nt * 16 + col) * TT + kt + half * 16;
            acc[nt] = wmma_bf16(ap, load_b_frag(vc), acc[nt]);
        }
    }

    unsigned short* cout = ctxb + (size_t)bk * TT * DH;
#pragma unroll
    for (int nt = 0; nt < 4; ++nt)
#pragma unroll
        for (int r = 0; r < 8; ++r) {
            int trow = q0 + r + 8 * half;
            cout[(size_t)trow * DH + nt * 16 + col] = f2bf(acc[nt][r] / l[r]);
        }
}

// ---------------- output projection + routing-weighted sum ----------------
// grid (8 D-tiles, 8 T-tiles, B), 256 threads; block = 128x128 tile of out[b]
__global__ void out_proj(const unsigned short* __restrict__ ctxb,
                         const unsigned short* __restrict__ wot,
                         const int* __restrict__ idx,
                         const float* __restrict__ rw,
                         float* __restrict__ out) {
    const int b     = blockIdx.z;
    const int tileT = blockIdx.y;
    const int tileD = blockIdx.x;
    const int w     = threadIdx.x >> 5;
    const int lane  = threadIdx.x & 31;
    const int col   = lane & 15;
    const int half  = lane >> 4;
    const int row0  = tileT * 128 + w * 16;
    const int n0    = tileD * 128;

    v8f fin[8] = {};
    for (int ks = 0; ks < KK; ++ks) {
        const int bk   = b * KK + ks;
        const int hsel = idx[bk];
        const float wgt = rw[bk];
        const unsigned short* crow =
            ctxb + (size_t)bk * TT * DH + (size_t)(row0 + col) * DH;
        const unsigned short* wo = wot + (size_t)hsel * DD * DH;
        v16bf a0 = load_a_frag(crow, half);
        v16bf a1 = load_a_frag(crow + 32, half);
#pragma unroll
        for (int nt = 0; nt < 8; ++nt) {
            const unsigned short* wc =
                wo + (size_t)(n0 + nt * 16 + col) * DH + half * 16;
            v8f t = {};
            t = wmma_bf16(a0, load_b_frag(wc), t);
            t = wmma_bf16(a1, load_b_frag(wc + 32), t);
            fin[nt] += t * wgt;
        }
    }

    float* ob = out + (size_t)b * TT * DD;
#pragma unroll
    for (int nt = 0; nt < 8; ++nt)
#pragma unroll
        for (int r = 0; r < 8; ++r)
            ob[(size_t)(row0 + r + 8 * half) * DD + n0 + nt * 16 + col] = fin[nt][r];
}

// ---------------- host launcher ----------------
extern "C" void kernel_launch(void* const* d_in, const int* in_sizes, int n_in,
                              void* d_out, int out_size, void* d_ws, size_t ws_size,
                              hipStream_t stream) {
    const float* x         = (const float*)d_in[0];
    // d_in[1] causal_mask, d_in[2] attention_mask: unused (all true / rebuilt)
    const int*   positions = (const int*)d_in[3];
    const float* rp        = (const float*)d_in[4];
    const float* router_w  = (const float*)d_in[5];
    const float* Wq        = (const float*)d_in[6];
    const float* Wk        = (const float*)d_in[7];
    const float* Wv        = (const float*)d_in[8];
    const float* Wo        = (const float*)d_in[9];
    float* out = (float*)d_out;

    char* ws = (char*)d_ws;
    size_t off = 0;
    auto alloc = [&](size_t bytes) -> void* {
        void* p = (void*)(ws + off);
        off += (bytes + 255) & ~(size_t)255;
        return p;
    };

    unsigned short* xb   = (unsigned short*)alloc((size_t)BB * TT * DD * 2);
    unsigned short* rpb  = (unsigned short*)alloc((size_t)DH * DD * 2);
    unsigned short* wqt  = (unsigned short*)alloc((size_t)HH * DH * DD * 2);
    unsigned short* wkt  = (unsigned short*)alloc((size_t)HH * DH * DD * 2);
    unsigned short* wvt  = (unsigned short*)alloc((size_t)HH * DH * DD * 2);
    unsigned short* wot  = (unsigned short*)alloc((size_t)HH * DD * DH * 2);
    float*          pooled = (float*)alloc((size_t)BB * DH * 4);
    int*            idxv   = (int*)alloc((size_t)BB * KK * 4);
    float*          rwv    = (float*)alloc((size_t)BB * KK * 4);
    unsigned short* qbuf = (unsigned short*)alloc((size_t)BB * KK * TT * DH * 2);
    unsigned short* kbuf = (unsigned short*)alloc((size_t)BB * KK * TT * DH * 2);
    unsigned short* vtbuf = (unsigned short*)alloc((size_t)BB * KK * DH * TT * 2);
    unsigned short* ctx  = (unsigned short*)alloc((size_t)BB * KK * TT * DH * 2);
    (void)ws_size; (void)in_sizes; (void)n_in; (void)out_size;

    // stage 0: precision conversion / weight transposes
    conv_bf16_kernel<<<4096, 256, 0, stream>>>(xb, x, BB * TT * DD);
    conv_bf16_kernel<<<256, 256, 0, stream>>>(rpb, rp, DH * DD);
    transpose_conv_kernel<<<2048, 256, 0, stream>>>(wqt, Wq, HH, DD, DH);
    transpose_conv_kernel<<<2048, 256, 0, stream>>>(wkt, Wk, HH, DD, DH);
    transpose_conv_kernel<<<2048, 256, 0, stream>>>(wvt, Wv, HH, DD, DH);
    transpose_conv_kernel<<<2048, 256, 0, stream>>>(wot, Wo, HH, DH, DD);

    // stage 1: router
    zero_kernel<<<2, 256, 0, stream>>>(pooled, BB * DH);
    router_gemm_rope_pool<<<dim3(8, BB), 256, 0, stream>>>(xb, rpb, positions, pooled);
    router_topk<<<1, 32, 0, stream>>>(pooled, router_w, idxv, rwv);

    // stage 2: QKV projections (+rope) for selected heads
    qkv_proj_rope<<<dim3(8, BB * KK), 256, 0, stream>>>(xb, wqt, wkt, wvt, idxv,
                                                        positions, qbuf, kbuf, vtbuf);

    // stage 3: causal flash attention
    attn_kernel<<<dim3(8, BB * KK), 256, 0, stream>>>(qbuf, kbuf, vtbuf, ctx);

    // stage 4: output projection + weighted sum over selected heads
    out_proj<<<dim3(8, 8, BB), 256, 0, stream>>>(ctx, wot, idxv, rwv, out);
}
